// HandWritingSynthesisNet_24000277250491
// MI455X (gfx1250) — compile-verified, matrix operand order checked by reading, not run on hardware
//
#include <hip/hip_runtime.h>

// ---------------- model constants ----------------
#define BB    64
#define TT    800
#define TXT   64
#define HH    400
#define EE    128
#define WW    10
#define OUTN  121
#define GATE  1600              // 4*H
#define KP1   544               // lstm1 K = 131+400=531 -> pad 544 (17*32)
#define KP23  960               // lstm2/3 K = 531+400=931 -> pad 960 (30*32)
#define KPO   1344              // out K = 1328 -> pad 1344 (42*32)
#define NOUTP 128               // out N = 121 -> pad 128

#define GRID_SCAN 40
#define NTH       256
#define NT_SCAN   (GRID_SCAN*NTH)   // 10240 threads
#define NW_SCAN   (NT_SCAN/32)      // 320 waves

#define LDS_BYTES (2*BB*KPO)        // 172032 B staging buffer (max of inbuf/outin)

// d_out layout (flat f32)
#define Y_OFF 0L
#define H_OFF 6195200L
#define C_OFF 6272000L
#define K_OFF 6348800L
#define W_OFF 6349440L

typedef __attribute__((ext_vector_type(16))) __bf16 v16bf;
typedef __attribute__((ext_vector_type(8)))  __bf16 v8bf;
typedef __attribute__((ext_vector_type(8)))  float  v8f;
struct V8x2 { v8bf lo, hi; };

__device__ inline unsigned short f2bf(float f) {
  unsigned u = __builtin_bit_cast(unsigned, f);
  unsigned r = (u + 0x7FFFu + ((u >> 16) & 1u)) >> 16;   // round-to-nearest-even
  return (unsigned short)r;
}
__device__ inline float bf2f(unsigned short b) {
  unsigned u = ((unsigned)b) << 16;
  return __builtin_bit_cast(float, u);
}
__device__ inline float sigm_(float x) { return 1.f / (1.f + __expf(-x)); }
__device__ inline float wred_sum(float v) { for (int m = 16; m > 0; m >>= 1) v += __shfl_xor(v, m, 32); return v; }
__device__ inline float wred_max(float v) { for (int m = 16; m > 0; m >>= 1) v = fmaxf(v, __shfl_xor(v, m, 32)); return v; }

// One 16x16 output tile of  C = A(MxK,bf16,row-major) * W(NxK,bf16,row-major)^T
// using v_wmma_f32_16x16x32_bf16, per the CDNA5 VGPR layouts.
// A may live in LDS (scan path) or global (encoder path); B streams from L2 with prefetch.
__device__ inline v8f wmma_tile_bf16(const unsigned short* A, int lda,
                                     const unsigned short* __restrict__ W, int ldw,
                                     int mt, int nt, int ktiles) {
  const int lane = threadIdx.x & 31;
  const int half = lane >> 4;
  const int idx  = lane & 15;
  const __bf16* arow = reinterpret_cast<const __bf16*>(A) + (size_t)(mt * 16 + idx) * lda;
  const __bf16* brow = reinterpret_cast<const __bf16*>(W) + (size_t)(nt * 16 + idx) * ldw;
  v8f acc = {};
  for (int kt = 0; kt < ktiles; ++kt) {
    const int k0 = kt * 32;
    v8bf a0 = *reinterpret_cast<const v8bf*>(arow + k0 + 8 * half);
    v8bf a1 = *reinterpret_cast<const v8bf*>(arow + k0 + 16 + 8 * half);
    v8bf b0 = *reinterpret_cast<const v8bf*>(brow + k0 + 8 * half);
    v8bf b1 = *reinterpret_cast<const v8bf*>(brow + k0 + 16 + 8 * half);
    if (kt + 1 < ktiles)                           // pull next weight K-slice toward L0/L2
      __builtin_prefetch(brow + k0 + 32, 0, 3);    // -> global_prefetch_b8
    v16bf a = __builtin_bit_cast(v16bf, V8x2{a0, a1});
    v16bf b = __builtin_bit_cast(v16bf, V8x2{b0, b1});
    acc = __builtin_amdgcn_wmma_f32_16x16x32_bf16(false, a, false, b, (short)0, acc, false, false);
  }
  return acc;
}

// ---------------- utility kernels ----------------
__global__ void __launch_bounds__(NTH) fill_bf0_kernel(unsigned short* p, long n) {
  for (long i = (long)blockIdx.x * blockDim.x + threadIdx.x; i < n; i += (long)gridDim.x * blockDim.x) p[i] = 0;
}
__global__ void __launch_bounds__(NTH) cvt_bf_kernel(const float* __restrict__ s, unsigned short* __restrict__ d, long n) {
  for (long i = (long)blockIdx.x * blockDim.x + threadIdx.x; i < n; i += (long)gridDim.x * blockDim.x) d[i] = f2bf(s[i]);
}
// dst[r*ldd + coff + k] = bf16(src[r*Ks + k])  (pack/concat weights with zero padding done separately)
__global__ void __launch_bounds__(NTH) cvt_region_kernel(const float* __restrict__ s, unsigned short* __restrict__ d,
                                                         int N, int Ks, int ldd, int coff) {
  long tot = (long)N * Ks;
  for (long i = (long)blockIdx.x * blockDim.x + threadIdx.x; i < tot; i += (long)gridDim.x * blockDim.x) {
    int r = (int)(i / Ks), k = (int)(i % Ks);
    d[(size_t)r * ldd + coff + k] = f2bf(s[i]);
  }
}

// ---------------- encoder kernels ----------------
__global__ void __launch_bounds__(NTH) embed_pos_kernel(const int* __restrict__ tok, const float* __restrict__ emb,
                                                        float* __restrict__ xf, unsigned short* __restrict__ xb) {
  const float sq = 11.313708498984761f;  // sqrt(128)
  for (long i = (long)blockIdx.x * blockDim.x + threadIdx.x; i < 4096L * 128; i += (long)gridDim.x * blockDim.x) {
    int row = (int)(i >> 7), e = (int)(i & 127);
    int l = row & 63;
    float div = __expf((float)(e & ~1) * (-9.210340371976184f / 128.f));
    float ang = (float)l * div;
    float pe = (e & 1) ? __cosf(ang) : __sinf(ang);
    float v = emb[tok[row] * EE + e] * sq + pe;
    xf[i] = v; xb[i] = f2bf(v);
  }
}

// generic GEMM: C = A @ W^T (+bias)(relu?) ; A:[Mx lda], W:[Nx ldw] bf16, writes f32 C and/or bf16 Cb
__global__ void __launch_bounds__(NTH) gemm_bf16_kernel(const unsigned short* __restrict__ A, int lda,
                                                        const unsigned short* __restrict__ W, int ldw,
                                                        float* C, int ldc, unsigned short* Cb, int ldcb,
                                                        const float* __restrict__ bias, int relu,
                                                        int M, int N, int ktiles) {
  const int gw = (blockIdx.x * blockDim.x + threadIdx.x) >> 5;
  const int nw = (gridDim.x * blockDim.x) >> 5;
  const int mtiles = M >> 4, ntiles = N >> 4, tot = mtiles * ntiles;
  const int lane = threadIdx.x & 31, half = lane >> 4, idx = lane & 15;
  for (int tile = gw; tile < tot; tile += nw) {
    const int mt = tile % mtiles, nt = tile / mtiles;
    v8f acc = wmma_tile_bf16(A, lda, W, ldw, mt, nt, ktiles);
    const int n = nt * 16 + idx;
    const float bv = bias ? bias[n] : 0.f;
    for (int r = 0; r < 8; ++r) {
      const int m = mt * 16 + 8 * half + r;
      float v = acc[r] + bv;
      if (relu) v = fmaxf(v, 0.f);
      if (C)  C[(size_t)m * ldc + n] = v;
      if (Cb) Cb[(size_t)m * ldcb + n] = f2bf(v);
    }
  }
}

// fused MHA: one wave per (b, head, qrow); dh=16, L=64
__global__ void __launch_bounds__(NTH) attn_kernel(const unsigned short* __restrict__ Q,
                                                   const unsigned short* __restrict__ Kb,
                                                   const unsigned short* __restrict__ Vb,
                                                   const float* __restrict__ mask,
                                                   unsigned short* __restrict__ Cb) {
  int gw = (blockIdx.x * blockDim.x + threadIdx.x) >> 5;
  if (gw >= 64 * 8 * 64) return;
  int lane = threadIdx.x & 31;
  int b = gw >> 9, h = (gw >> 6) & 7, qi = gw & 63;
  const unsigned short* qp = Q + ((size_t)(b * 64 + qi) * EE + h * 16);
  float qv[16];
  for (int e = 0; e < 16; ++e) qv[e] = bf2f(qp[e]);
  float s[2];
  for (int rep = 0; rep < 2; ++rep) {
    int k = lane + 32 * rep;
    const unsigned short* kp = Kb + ((size_t)(b * 64 + k) * EE + h * 16);
    float acc = 0.f;
    for (int e = 0; e < 16; ++e) acc += qv[e] * bf2f(kp[e]);
    acc *= 0.25f;                                   // 1/sqrt(16)
    if (mask[b * 64 + k] == 0.f) acc = -1e9f;
    s[rep] = acc;
  }
  float mx = wred_max(fmaxf(s[0], s[1]));
  float p0 = __expf(s[0] - mx), p1 = __expf(s[1] - mx);
  float sum = wred_sum(p0 + p1);
  float a0 = p0 / sum, a1 = p1 / sum;
  const unsigned short* v0 = Vb + ((size_t)(b * 64 + lane) * EE + h * 16);
  const unsigned short* v1 = v0 + 32 * EE;
  unsigned short* op = Cb + ((size_t)(b * 64 + qi) * EE + h * 16);
  for (int e = 0; e < 16; ++e) {
    float c = wred_sum(a0 * bf2f(v0[e]) + a1 * bf2f(v1[e]));
    if (lane == 0) op[e] = f2bf(c);
  }
}

// out = LN(X + R)*g + b ; one wave per row of 128
__global__ void __launch_bounds__(NTH) add_ln_kernel(const float* __restrict__ X, const float* __restrict__ R,
                                                     const float* __restrict__ g, const float* __restrict__ bt,
                                                     float* __restrict__ Of, unsigned short* __restrict__ Ob) {
  int gw = (blockIdx.x * blockDim.x + threadIdx.x) >> 5;
  if (gw >= 4096) return;
  int lane = threadIdx.x & 31;
  float v[4];
  for (int j = 0; j < 4; ++j) { int e = lane + 32 * j; v[j] = X[(size_t)gw * EE + e] + R[(size_t)gw * EE + e]; }
  float mean = wred_sum(v[0] + v[1] + v[2] + v[3]) * (1.f / 128.f);
  float d0 = v[0] - mean, d1 = v[1] - mean, d2 = v[2] - mean, d3 = v[3] - mean;
  float var = wred_sum(d0 * d0 + d1 * d1 + d2 * d2 + d3 * d3) * (1.f / 128.f);
  float inv = rsqrtf(var + 1e-5f);
  for (int j = 0; j < 4; ++j) {
    int e = lane + 32 * j;
    float o = (v[j] - mean) * inv * g[e] + bt[e];
    Of[(size_t)gw * EE + e] = o; Ob[(size_t)gw * EE + e] = f2bf(o);
  }
}

// ---------------- scan init ----------------
__global__ void __launch_bounds__(NTH) scan_init_kernel(
    const float* h0, const float* c0, const float* k0,
    const float* bih1, const float* bhh1, const float* bih2, const float* bhh2,
    const float* bih3, const float* bhh3,
    float* h1f, float* h2f, float* h3f, float* c1f, float* c2f, float* c3f,
    float* kap, float* bs1, float* bs2, float* bs3,
    unsigned short* inbuf, unsigned short* outin, unsigned* bar) {
  long tid = (long)blockIdx.x * blockDim.x + threadIdx.x;
  long nt = (long)gridDim.x * blockDim.x;
  for (long i = tid; i < 25600; i += nt) {
    h1f[i] = h0[i]; h2f[i] = h0[25600 + i]; h3f[i] = h0[51200 + i];
    c1f[i] = c0[i]; c2f[i] = c0[25600 + i]; c3f[i] = c0[51200 + i];
    int b = (int)(i / HH), u = (int)(i % HH);
    inbuf[b * KP23 + 131 + u] = f2bf(h0[i]);   // hA = h1_prev
    inbuf[b * KP23 + 531 + u] = 0;             // hB
  }
  for (long i = tid; i < GATE; i += nt) { bs1[i] = bih1[i] + bhh1[i]; bs2[i] = bih2[i] + bhh2[i]; bs3[i] = bih3[i] + bhh3[i]; }
  for (long i = tid; i < BB * WW; i += nt) kap[i] = k0[i];
  for (long i = tid; i < 64L * 29; i += nt) { int b = (int)(i / 29), k = (int)(i % 29); inbuf[b * KP23 + 931 + k] = 0; }
  for (long i = tid; i < 64L * 16; i += nt) { int b = (int)(i / 16), k = (int)(i % 16); outin[b * KPO + 1328 + k] = 0; }
  for (long i = tid; i < 128; i += nt) bar[i] = 0;
}

// ---------------- persistent scan kernel ----------------
__device__ inline void gbar(unsigned* cnt, unsigned* ph) {
  __threadfence();
  __syncthreads();
  if (threadIdx.x == 0) {
    unsigned cur = __hip_atomic_load(ph, __ATOMIC_ACQUIRE, __HIP_MEMORY_SCOPE_AGENT);
    unsigned a = __hip_atomic_fetch_add(cnt, 1u, __ATOMIC_ACQ_REL, __HIP_MEMORY_SCOPE_AGENT);
    if (a == (unsigned)(GRID_SCAN - 1)) {
      __hip_atomic_store(cnt, 0u, __ATOMIC_RELAXED, __HIP_MEMORY_SCOPE_AGENT);
      __hip_atomic_fetch_add(ph, 1u, __ATOMIC_RELEASE, __HIP_MEMORY_SCOPE_AGENT);
    } else {
      while (__hip_atomic_load(ph, __ATOMIC_ACQUIRE, __HIP_MEMORY_SCOPE_AGENT) == cur)
        __builtin_amdgcn_s_sleep(2);
    }
  }
  __syncthreads();
}

// Stage a block of global memory into the dynamic-LDS segment via the CDNA5
// async copy path (global_load_async_to_lds_b128, tracked by ASYNCcnt).
// Dynamic LDS starts at groupstaticsize(); each thread copies 16B chunks.
__device__ inline void stage_to_lds_async(const void* gsrc, unsigned bytes) {
  const unsigned base = __builtin_amdgcn_groupstaticsize();
  const char* g = (const char*)gsrc;
  for (unsigned off = (unsigned)threadIdx.x * 16u; off < bytes; off += NTH * 16u) {
    unsigned ldsaddr = base + off;
    const void* gp = g + off;
    asm volatile("global_load_async_to_lds_b128 %0, %1, off"
                 :: "v"(ldsaddr), "v"(gp) : "memory");
  }
  asm volatile("s_wait_asynccnt 0x0" ::: "memory");   // our wave's copies landed
  __syncthreads();                                    // whole WG's copies landed
}

__global__ void __launch_bounds__(NTH, 1) scan_kernel(
    const float* __restrict__ stroke, const float* __restrict__ mask,
    const float* __restrict__ Wa, const float* __restrict__ ba,
    const float* __restrict__ enc,
    const unsigned short* __restrict__ Wc1, const unsigned short* __restrict__ Wc2,
    const unsigned short* __restrict__ Wc3,
    const float* __restrict__ bs1, const float* __restrict__ bs2, const float* __restrict__ bs3,
    const unsigned short* __restrict__ Wob, const float* __restrict__ bout,
    float* h1f, float* h2f, float* h3f, float* c1f, float* c2f, float* c3f,
    float* kap, float* alp, float* bet, float* phi, float* phisum, float* wtf,
    unsigned short* inbuf, unsigned short* outin, float* gates,
    float* yout, unsigned* bar) {
  extern __shared__ unsigned short smem[];            // LDS_BYTES staging buffer
  const int tid = blockIdx.x * blockDim.x + threadIdx.x;
  const int gw = tid >> 5;
  const int lane = threadIdx.x & 31;
  const int half = lane >> 4, idx = lane & 15;
  unsigned* cnt = bar;
  unsigned* ph = bar + 64;

#pragma unroll 1
  for (int t = 0; t < TT; ++t) {
    // --- A1: window params wp = h1 @ Wa^T + ba ; kappa += exp(k_hat); stage x_t
    for (int i = tid; i < BB * 30; i += NT_SCAN) {
      int b = i / 30, j = i % 30;
      const float* hr = h1f + b * HH;
      const float* wr = Wa + j * HH;
      float s = ba[j];
      for (int k = 0; k < HH; ++k) s += hr[k] * wr[k];
      float e = __expf(s);
      if (j < 10) alp[b * WW + j] = e;
      else if (j < 20) bet[b * WW + (j - 10)] = e;
      else kap[b * WW + (j - 20)] += e;
    }
    for (int i = tid; i < BB * 3; i += NT_SCAN) {
      int b = i / 3, k = i % 3;
      inbuf[b * KP23 + k] = f2bf(stroke[((size_t)b * TT + t) * 3 + k]);
    }
    gbar(cnt, ph);
    // --- A2: phi + per-b sum (wave b handles chars lane, lane+32)
    if (gw < BB) {
      int b = gw;
      float ps = 0.f;
      for (int rep = 0; rep < 2; ++rep) {
        int l = lane + 32 * rep;
        float u = (float)l, acc = 0.f;
        for (int w = 0; w < WW; ++w) {
          float d = kap[b * WW + w] - u;
          acc += alp[b * WW + w] * __expf(-bet[b * WW + w] * d * d);
        }
        acc *= mask[b * TXT + l];
        phi[b * TXT + l] = acc;
        ps += acc;
      }
      ps = wred_sum(ps);
      if (lane == 0) phisum[b] = ps + 1e-8f;
    }
    gbar(cnt, ph);
    // --- A3: w_t = phi_n @ text_enc ; stage into inbuf/outin
    for (int i = tid; i < BB * EE; i += NT_SCAN) {
      int b = i >> 7, e = i & 127;
      float acc = 0.f;
      const float* pr = phi + b * TXT;
      const float* er = enc + (size_t)b * TXT * EE + e;
      for (int l = 0; l < TXT; ++l) acc += pr[l] * er[(size_t)l * EE];
      acc /= phisum[b];
      wtf[i] = acc;
      unsigned short bv = f2bf(acc);
      inbuf[b * KP23 + 3 + e] = bv;
      outin[b * KPO + 1200 + e] = bv;
    }
    gbar(cnt, ph);
    // --- B: gates1 = [x,w,h1] @ [Wih1|Whh1]^T   (A from LDS, WMMA bf16, K=544)
    stage_to_lds_async(inbuf, 2u * BB * KP23);
    for (int tile = gw; tile < 400; tile += NW_SCAN) {
      int nt = tile >> 2, mt = tile & 3;
      v8f acc = wmma_tile_bf16(smem, KP23, Wc1, KP1, mt, nt, KP1 / 32);
      int n = nt * 16 + idx;
      for (int r = 0; r < 8; ++r) gates[(size_t)(mt * 16 + 8 * half + r) * GATE + n] = acc[r];
    }
    gbar(cnt, ph);
    // --- C: lstm1 pointwise ; hA<-h1_new, hB<-h2_prev
    for (int i = tid; i < BB * HH; i += NT_SCAN) {
      int b = i / HH, u = i - b * HH;
      const float* g = gates + (size_t)b * GATE;
      float ig = g[u] + bs1[u], fg = g[400 + u] + bs1[400 + u];
      float gg = g[800 + u] + bs1[800 + u], og = g[1200 + u] + bs1[1200 + u];
      float c = sigm_(fg) * c1f[i] + sigm_(ig) * tanhf(gg);
      float h = sigm_(og) * tanhf(c);
      c1f[i] = c; h1f[i] = h;
      unsigned short hb = f2bf(h);
      inbuf[b * KP23 + 131 + u] = hb;
      outin[b * KPO + u] = hb;
      inbuf[b * KP23 + 531 + u] = f2bf(h2f[i]);
    }
    gbar(cnt, ph);
    // --- D: gates2 (A from LDS, K=960)
    stage_to_lds_async(inbuf, 2u * BB * KP23);
    for (int tile = gw; tile < 400; tile += NW_SCAN) {
      int nt = tile >> 2, mt = tile & 3;
      v8f acc = wmma_tile_bf16(smem, KP23, Wc2, KP23, mt, nt, KP23 / 32);
      int n = nt * 16 + idx;
      for (int r = 0; r < 8; ++r) gates[(size_t)(mt * 16 + 8 * half + r) * GATE + n] = acc[r];
    }
    gbar(cnt, ph);
    // --- E: lstm2 pointwise ; hA<-h2_new, hB<-h3_prev
    for (int i = tid; i < BB * HH; i += NT_SCAN) {
      int b = i / HH, u = i - b * HH;
      const float* g = gates + (size_t)b * GATE;
      float ig = g[u] + bs2[u], fg = g[400 + u] + bs2[400 + u];
      float gg = g[800 + u] + bs2[800 + u], og = g[1200 + u] + bs2[1200 + u];
      float c = sigm_(fg) * c2f[i] + sigm_(ig) * tanhf(gg);
      float h = sigm_(og) * tanhf(c);
      c2f[i] = c; h2f[i] = h;
      unsigned short hb = f2bf(h);
      inbuf[b * KP23 + 131 + u] = hb;
      outin[b * KPO + 400 + u] = hb;
      inbuf[b * KP23 + 531 + u] = f2bf(h3f[i]);
    }
    gbar(cnt, ph);
    // --- F: gates3 (A from LDS, K=960)
    stage_to_lds_async(inbuf, 2u * BB * KP23);
    for (int tile = gw; tile < 400; tile += NW_SCAN) {
      int nt = tile >> 2, mt = tile & 3;
      v8f acc = wmma_tile_bf16(smem, KP23, Wc3, KP23, mt, nt, KP23 / 32);
      int n = nt * 16 + idx;
      for (int r = 0; r < 8; ++r) gates[(size_t)(mt * 16 + 8 * half + r) * GATE + n] = acc[r];
    }
    gbar(cnt, ph);
    // --- G: lstm3 pointwise ; hA<-h1(current) for next step's gates1
    for (int i = tid; i < BB * HH; i += NT_SCAN) {
      int b = i / HH, u = i - b * HH;
      const float* g = gates + (size_t)b * GATE;
      float ig = g[u] + bs3[u], fg = g[400 + u] + bs3[400 + u];
      float gg = g[800 + u] + bs3[800 + u], og = g[1200 + u] + bs3[1200 + u];
      float c = sigm_(fg) * c3f[i] + sigm_(ig) * tanhf(gg);
      float h = sigm_(og) * tanhf(c);
      c3f[i] = c; h3f[i] = h;
      outin[b * KPO + 800 + u] = f2bf(h);
      inbuf[b * KP23 + 131 + u] = f2bf(h1f[i]);
    }
    gbar(cnt, ph);
    // --- H: y_t = [h1,h2,h3,w] @ Wout^T + bout  (A from LDS, K=1344, N padded to 128)
    stage_to_lds_async(outin, 2u * BB * KPO);
    for (int tile = gw; tile < 32; tile += NW_SCAN) {
      int nt = tile >> 2, mt = tile & 3;
      v8f acc = wmma_tile_bf16(smem, KPO, Wob, KPO, mt, nt, KPO / 32);
      int n = nt * 16 + idx;
      if (n < OUTN) {
        for (int r = 0; r < 8; ++r) {
          int b = mt * 16 + 8 * half + r;
          yout[((size_t)b * TT + t) * OUTN + n] = acc[r] + bout[n];
        }
      }
    }
    gbar(cnt, ph);
  }
  // final state outputs
  for (int i = tid; i < 25600; i += NT_SCAN) {
    yout[H_OFF + i] = h1f[i]; yout[H_OFF + 25600 + i] = h2f[i]; yout[H_OFF + 51200 + i] = h3f[i];
    yout[C_OFF + i] = c1f[i]; yout[C_OFF + 25600 + i] = c2f[i]; yout[C_OFF + 51200 + i] = c3f[i];
  }
  for (int i = tid; i < BB * WW; i += NT_SCAN) yout[K_OFF + i] = kap[i];
  for (int i = tid; i < BB * EE; i += NT_SCAN) yout[W_OFF + i] = wtf[i];
}

// ---------------- host ----------------
extern "C" void kernel_launch(void* const* d_in, const int* in_sizes, int n_in,
                              void* d_out, int out_size, void* d_ws, size_t ws_size,
                              hipStream_t stream) {
  (void)in_sizes; (void)n_in; (void)out_size; (void)ws_size;
  const float* stroke = (const float*)d_in[0];
  const int*   tokens = (const int*)d_in[1];
  const float* mask   = (const float*)d_in[2];
  const float* h0     = (const float*)d_in[3];
  const float* c0     = (const float*)d_in[4];
  const float* kappa0 = (const float*)d_in[5];
  const float* emb    = (const float*)d_in[6];
  const float* Wa  = (const float*)d_in[39];
  const float* ba  = (const float*)d_in[40];
  const float* Wih1 = (const float*)d_in[41]; const float* Whh1 = (const float*)d_in[42];
  const float* bih1 = (const float*)d_in[43]; const float* bhh1 = (const float*)d_in[44];
  const float* Wih2 = (const float*)d_in[45]; const float* Whh2 = (const float*)d_in[46];
  const float* bih2 = (const float*)d_in[47]; const float* bhh2 = (const float*)d_in[48];
  const float* Wih3 = (const float*)d_in[49]; const float* Whh3 = (const float*)d_in[50];
  const float* bih3 = (const float*)d_in[51]; const float* bhh3 = (const float*)d_in[52];
  const float* Wout = (const float*)d_in[53]; const float* bout = (const float*)d_in[54];

  char* base = (char*)d_ws;
  size_t off = 0;
  auto alloc = [&](size_t bytes) -> char* {
    off = (off + 255) & ~(size_t)255;
    char* p = base + off; off += bytes; return p;
  };
  // packed bf16 weights
  unsigned short* Wc1 = (unsigned short*)alloc(2L * GATE * KP1);
  unsigned short* Wc2 = (unsigned short*)alloc(2L * GATE * KP23);
  unsigned short* Wc3 = (unsigned short*)alloc(2L * GATE * KP23);
  unsigned short* Wob = (unsigned short*)alloc(2L * NOUTP * KPO);
  unsigned short* encW[2][6];
  static const long encWn[6] = {EE * EE, EE * EE, EE * EE, EE * EE, 800L * EE, EE * 800L};
  for (int l = 0; l < 2; ++l)
    for (int k = 0; k < 6; ++k) encW[l][k] = (unsigned short*)alloc(2L * encWn[k]);
  float* bs1 = (float*)alloc(4L * GATE);
  float* bs2 = (float*)alloc(4L * GATE);
  float* bs3 = (float*)alloc(4L * GATE);
  // encoder activations (rows = b*64+l)
  float* xf  = (float*)alloc(4L * 4096 * EE);  unsigned short* xb  = (unsigned short*)alloc(2L * 4096 * EE);
  unsigned short* qb = (unsigned short*)alloc(2L * 4096 * EE);
  unsigned short* kb = (unsigned short*)alloc(2L * 4096 * EE);
  unsigned short* vb = (unsigned short*)alloc(2L * 4096 * EE);
  unsigned short* cb = (unsigned short*)alloc(2L * 4096 * EE);
  float* of  = (float*)alloc(4L * 4096 * EE);
  float* x1f = (float*)alloc(4L * 4096 * EE);  unsigned short* x1b = (unsigned short*)alloc(2L * 4096 * EE);
  float* fff = (float*)alloc(4L * 4096 * 800); unsigned short* ffb = (unsigned short*)alloc(2L * 4096 * 800);
  float* ff2 = (float*)alloc(4L * 4096 * EE);
  // scan state
  float* h1f = (float*)alloc(4L * 25600); float* h2f = (float*)alloc(4L * 25600); float* h3f = (float*)alloc(4L * 25600);
  float* c1f = (float*)alloc(4L * 25600); float* c2f = (float*)alloc(4L * 25600); float* c3f = (float*)alloc(4L * 25600);
  float* kap = (float*)alloc(4L * BB * WW);
  float* alp = (float*)alloc(4L * BB * WW);
  float* bet = (float*)alloc(4L * BB * WW);
  float* phi = (float*)alloc(4L * BB * TXT);
  float* phs = (float*)alloc(4L * BB);
  float* wtf = (float*)alloc(4L * BB * EE);
  unsigned short* inbuf = (unsigned short*)alloc(2L * BB * KP23);
  unsigned short* outin = (unsigned short*)alloc(2L * BB * KPO);
  float* gates = (float*)alloc(4L * BB * GATE);
  unsigned* bar = (unsigned*)alloc(4L * 128);

  // ---- weight packing (bf16, zero-padded / concatenated [Wih|Whh|0]) ----
  fill_bf0_kernel<<<512, NTH, 0, stream>>>(Wc1, (long)GATE * KP1);
  fill_bf0_kernel<<<512, NTH, 0, stream>>>(Wc2, (long)GATE * KP23);
  fill_bf0_kernel<<<512, NTH, 0, stream>>>(Wc3, (long)GATE * KP23);
  fill_bf0_kernel<<<128, NTH, 0, stream>>>(Wob, (long)NOUTP * KPO);
  cvt_region_kernel<<<512, NTH, 0, stream>>>(Wih1, Wc1, GATE, 131, KP1, 0);
  cvt_region_kernel<<<512, NTH, 0, stream>>>(Whh1, Wc1, GATE, 400, KP1, 131);
  cvt_region_kernel<<<512, NTH, 0, stream>>>(Wih2, Wc2, GATE, 531, KP23, 0);
  cvt_region_kernel<<<512, NTH, 0, stream>>>(Whh2, Wc2, GATE, 400, KP23, 531);
  cvt_region_kernel<<<512, NTH, 0, stream>>>(Wih3, Wc3, GATE, 531, KP23, 0);
  cvt_region_kernel<<<512, NTH, 0, stream>>>(Whh3, Wc3, GATE, 400, KP23, 531);
  cvt_region_kernel<<<256, NTH, 0, stream>>>(Wout, Wob, OUTN, 1328, KPO, 0);
  static const int encWidx[6] = {0, 2, 4, 6, 10, 12};  // Wq,Wk,Wv,Wo,W1,W2 within layer block
  for (int l = 0; l < 2; ++l)
    for (int k = 0; k < 6; ++k)
      cvt_bf_kernel<<<256, NTH, 0, stream>>>((const float*)d_in[7 + l * 16 + encWidx[k]], encW[l][k], encWn[k]);

  // ---- text encoder ----
  embed_pos_kernel<<<1024, NTH, 0, stream>>>(tokens, emb, xf, xb);
  for (int l = 0; l < 2; ++l) {
    const float* bq = (const float*)d_in[7 + l * 16 + 1];
    const float* bk = (const float*)d_in[7 + l * 16 + 3];
    const float* bv = (const float*)d_in[7 + l * 16 + 5];
    const float* bo = (const float*)d_in[7 + l * 16 + 7];
    const float* g1 = (const float*)d_in[7 + l * 16 + 8];
    const float* b1n = (const float*)d_in[7 + l * 16 + 9];
    const float* bf1 = (const float*)d_in[7 + l * 16 + 11];
    const float* bf2 = (const float*)d_in[7 + l * 16 + 13];
    const float* g2 = (const float*)d_in[7 + l * 16 + 14];
    const float* b2n = (const float*)d_in[7 + l * 16 + 15];
    gemm_bf16_kernel<<<256, NTH, 0, stream>>>(xb, EE, encW[l][0], EE, nullptr, 0, qb, EE, bq, 0, 4096, EE, 4);
    gemm_bf16_kernel<<<256, NTH, 0, stream>>>(xb, EE, encW[l][1], EE, nullptr, 0, kb, EE, bk, 0, 4096, EE, 4);
    gemm_bf16_kernel<<<256, NTH, 0, stream>>>(xb, EE, encW[l][2], EE, nullptr, 0, vb, EE, bv, 0, 4096, EE, 4);
    attn_kernel<<<4096, NTH, 0, stream>>>(qb, kb, vb, mask, cb);
    gemm_bf16_kernel<<<256, NTH, 0, stream>>>(cb, EE, encW[l][3], EE, of, EE, nullptr, 0, bo, 0, 4096, EE, 4);
    add_ln_kernel<<<512, NTH, 0, stream>>>(xf, of, g1, b1n, x1f, x1b);
    gemm_bf16_kernel<<<1024, NTH, 0, stream>>>(x1b, EE, encW[l][4], EE, fff, 800, ffb, 800, bf1, 1, 4096, 800, 4);
    gemm_bf16_kernel<<<256, NTH, 0, stream>>>(ffb, 800, encW[l][5], 800, ff2, EE, nullptr, 0, bf2, 0, 4096, EE, 25);
    add_ln_kernel<<<512, NTH, 0, stream>>>(x1f, ff2, g2, b2n, xf, xb);
  }

  // ---- recurrent scan (persistent kernel, LDS-staged A operand, grid barriers) ----
  hipFuncSetAttribute(reinterpret_cast<const void*>(scan_kernel),
                      hipFuncAttributeMaxDynamicSharedMemorySize, LDS_BYTES);
  scan_init_kernel<<<256, NTH, 0, stream>>>(h0, c0, kappa0,
      bih1, bhh1, bih2, bhh2, bih3, bhh3,
      h1f, h2f, h3f, c1f, c2f, c3f, kap, bs1, bs2, bs3, inbuf, outin, bar);
  scan_kernel<<<GRID_SCAN, NTH, LDS_BYTES, stream>>>(stroke, mask, Wa, ba, xf,
      Wc1, Wc2, Wc3, bs1, bs2, bs3, Wob, bout,
      h1f, h2f, h3f, c1f, c2f, c3f,
      kap, alp, bet, phi, phs, wtf, inbuf, outin, gates,
      (float*)d_out, bar);
}
